// DG2N_71906342469742
// MI455X (gfx1250) — compile-verified, matrix operand order checked by reading, not run on hardware
//
#include <hip/hip_runtime.h>
#include <hip/hip_bf16.h>

// GNN forward for MI455X (gfx1250).
// - All GEMMs via v_wmma_f32_16x16x32_bf16 (bf16 in, fp32 accumulate).
// - Weights pre-converted once per launch to transposed bf16 [N][K] so B tiles
//   are staged with GLOBAL_LOAD_ASYNC_TO_LDS_B128 (ASYNCcnt path), no VGPR trip.
// - Gather (edge_src), feature gate, K-split concat, edge gate and segment-sum
//   scatter-add all fused into one generic GEMM kernel.

typedef __attribute__((ext_vector_type(16))) __bf16 v16bf;
typedef __attribute__((ext_vector_type(8)))  float  v8f;
typedef int v4i __attribute__((vector_size(4 * sizeof(int))));
typedef __attribute__((address_space(1))) v4i* gas_v4i_ptr;   // __device__ (global)
typedef __attribute__((address_space(3))) v4i* las_v4i_ptr;   // __shared__ (LDS)

#define BM 128
#define BN 128
#define BK 32
#define LDAH 40   // halves per As row (BK + 8 pad)
#define LDBH 40   // halves per Bs row  (BK + 8 pad)

#if defined(__has_builtin)
#  if __has_builtin(__builtin_amdgcn_global_load_async_to_lds_b128) && \
      __has_builtin(__builtin_amdgcn_s_wait_asynccnt)
#    define USE_ASYNC_LDS 1
#  endif
#endif
#ifndef USE_ASYNC_LDS
#  define USE_ASYNC_LDS 0
#endif

union Frag {
  uint4  u[2];
  v16bf  v;
};

__device__ __forceinline__ unsigned short f2bf(float f) {
  unsigned u = __float_as_uint(f);
  unsigned r = u + 0x7FFFu + ((u >> 16) & 1u);   // round-to-nearest-even
  return (unsigned short)(r >> 16);
}

__global__ void zero_f32(float* __restrict__ p, long n) {
  long i = (long)blockIdx.x * blockDim.x + threadIdx.x;
  long stride = (long)gridDim.x * blockDim.x;
  for (; i < n; i += stride) p[i] = 0.f;
}

// W fp32 [K,N] row-major  ->  Wt bf16 [N,K] row-major (transposed)
__global__ void conv_transpose_bf16(const float* __restrict__ src,
                                    unsigned short* __restrict__ dst,
                                    int K, int N) {
  int i = blockIdx.x * blockDim.x + threadIdx.x;
  if (i < K * N) {
    int k = i / N, n = i - k * N;
    dst[(size_t)n * K + k] = f2bf(src[i]);
  }
}

// C[M,N] = epilogue( A[M,K] @ W[K,N] + bias ), W given as bf16-transposed Wt[N,K].
__global__ __launch_bounds__(256)
void gnn_wmma_gemm(const float* __restrict__ A0, int lda0,
                   const float* __restrict__ A1, int lda1, int K0,
                   const int* __restrict__ gatherIdx,
                   const float* __restrict__ colGate,
                   const unsigned short* __restrict__ Wt,   // bf16 [Nn][K]
                   const float* __restrict__ bias,
                   const float* __restrict__ gatePtr,
                   float* __restrict__ Cout, int ldc,
                   const int* __restrict__ scatterIdx,
                   int M, int Nn, int K, int doRelu)
{
  __shared__ __align__(16) unsigned short As[BM * LDAH];
  __shared__ __align__(16) unsigned short Bs[BN * LDBH];

  const int tid  = threadIdx.x;
  const int lane = tid & 31;
  const int wv   = tid >> 5;
  const int wM   = wv & 3;          // 4 waves along M (32 rows each)
  const int wN   = wv >> 2;         // 2 waves along N (64 cols each)
  const int blockRow = blockIdx.y * BM;
  const int blockCol = blockIdx.x * BN;

  v8f acc[2][4];
  #pragma unroll
  for (int i = 0; i < 2; ++i)
    #pragma unroll
    for (int j = 0; j < 4; ++j)
      #pragma unroll
      for (int r = 0; r < 8; ++r) acc[i][j][r] = 0.f;

  // A-tile load map: rows {tid>>2, tid>>2+64}, cols (tid&3)*8..+8
  const int aCol  = (tid & 3) * 8;
  const int aRow0 = tid >> 2;

  int srcRow[2];
  #pragma unroll
  for (int r = 0; r < 2; ++r) {
    int gr = blockRow + aRow0 + r * 64;
    srcRow[r] = (gr < M) ? (gatherIdx ? gatherIdx[gr] : gr) : -1;
  }

  for (int k0 = 0; k0 < K; k0 += BK) {
    // ---- stage B tile: bf16 rows, 16B per lane, async global->LDS ----
    #pragma unroll
    for (int s = 0; s < 2; ++s) {
      int seg = tid * 2 + s;            // 512 x 16B segments = 8KB tile
      int n   = seg >> 2;               // 0..127 (B row = output column)
      int ks  = (seg & 3) * 8;          // 0,8,16,24 (halves within K chunk)
      int gn  = blockCol + n;
      if (gn < Nn) {
        const unsigned short* gsrc = Wt + (size_t)gn * K + (k0 + ks);
        unsigned short* ldst = &Bs[n * LDBH + ks];
#if USE_ASYNC_LDS
        __builtin_amdgcn_global_load_async_to_lds_b128(
            (gas_v4i_ptr)(void*)gsrc, (las_v4i_ptr)(void*)ldst, 0, 0);
#else
        *(uint4*)ldst = *(const uint4*)gsrc;
#endif
      }
    }
    // ---- stage A tile (fp32 -> bf16, optional gather + column gate) ----
    #pragma unroll
    for (int r = 0; r < 2; ++r) {
      int row = aRow0 + r * 64;
      unsigned short* dst = &As[row * LDAH + aCol];
      float vals[8];
      int sr = srcRow[r];
      if (sr >= 0) {
        int gk = k0 + aCol;    // K0 is a multiple of BK -> no straddle
        const float* src = (gk < K0) ? (A0 + (size_t)sr * lda0 + gk)
                                     : (A1 + (size_t)sr * lda1 + (gk - K0));
        float4 v0 = *(const float4*)(src);
        float4 v1 = *(const float4*)(src + 4);
        vals[0] = v0.x; vals[1] = v0.y; vals[2] = v0.z; vals[3] = v0.w;
        vals[4] = v1.x; vals[5] = v1.y; vals[6] = v1.z; vals[7] = v1.w;
        if (colGate) {
          #pragma unroll
          for (int j = 0; j < 8; ++j)
            vals[j] = (colGate[gk + j] >= 0.f) ? vals[j] : 0.f;
        }
      } else {
        #pragma unroll
        for (int j = 0; j < 8; ++j) vals[j] = 0.f;
      }
      #pragma unroll
      for (int j = 0; j < 8; ++j) dst[j] = f2bf(vals[j]);
    }
#if USE_ASYNC_LDS
    __builtin_amdgcn_s_wait_asynccnt(0);
#endif
    __syncthreads();

    // ---- fragments per ISA 16-bit layout ----
    // lanes 0-15: halves 0..7 = K kb..kb+7, halves 8..15 = K kb+16..kb+23
    const int mrow = lane & 15;
    const int kb   = (lane < 16) ? 0 : 8;
    Frag af[2], bf[4];
    #pragma unroll
    for (int mt = 0; mt < 2; ++mt) {
      int rA = (wM * 32 + mt * 16 + mrow) * LDAH + kb;
      af[mt].u[0] = *(const uint4*)&As[rA];
      af[mt].u[1] = *(const uint4*)&As[rA + 16];
    }
    #pragma unroll
    for (int nt = 0; nt < 4; ++nt) {
      int rB = (wN * 64 + nt * 16 + mrow) * LDBH + kb;
      bf[nt].u[0] = *(const uint4*)&Bs[rB];
      bf[nt].u[1] = *(const uint4*)&Bs[rB + 16];
    }
    #pragma unroll
    for (int mt = 0; mt < 2; ++mt)
      #pragma unroll
      for (int nt = 0; nt < 4; ++nt)
        acc[mt][nt] = __builtin_amdgcn_wmma_f32_16x16x32_bf16(
            false, af[mt].v, false, bf[nt].v,
            (short)0, acc[mt][nt], false, false);
    __syncthreads();
  }

  float gate = 1.f;
  if (gatePtr) gate = (gatePtr[0] >= 0.f) ? 1.f : 0.f;

  // C/D layout: VGPR r -> row r (lanes 0-15) or r+8 (lanes 16-31), col=lane&15
  #pragma unroll
  for (int mt = 0; mt < 2; ++mt) {
    int rowBase = blockRow + wM * 32 + mt * 16 + ((lane & 16) ? 8 : 0);
    #pragma unroll
    for (int nt = 0; nt < 4; ++nt) {
      int col = blockCol + wN * 64 + nt * 16 + (lane & 15);
      if (col < Nn) {
        float bv = bias ? bias[col] : 0.f;
        #pragma unroll
        for (int r = 0; r < 8; ++r) {
          int row = rowBase + r;
          if (row < M) {
            float v = acc[mt][nt][r] + bv;
            if (doRelu) v = fmaxf(v, 0.f);
            v *= gate;
            if (scatterIdx) {
              int drow = scatterIdx[row];
              atomicAdd(&Cout[(size_t)drow * ldc + col], v);
            } else {
              Cout[(size_t)row * ldc + col] = v;
            }
          }
        }
      }
    }
  }
}

extern "C" void kernel_launch(void* const* d_in, const int* in_sizes, int n_in,
                              void* d_out, int out_size, void* d_ws, size_t ws_size,
                              hipStream_t stream) {
  (void)in_sizes; (void)n_in; (void)out_size; (void)ws_size;

  const float* x        = (const float*)d_in[0];
  const int*   edge_src = (const int*)d_in[1];
  const int*   edge_dst = (const int*)d_in[2];
  const float* feat_la  = (const float*)d_in[3];
  const float* edge_la  = (const float*)d_in[4];
  const float* in_w     = (const float*)d_in[5];
  const float* in_b     = (const float*)d_in[6];
  const float* msg_w1   = (const float*)d_in[7];
  const float* msg_b1   = (const float*)d_in[8];
  const float* msg_w2   = (const float*)d_in[9];
  const float* msg_b2   = (const float*)d_in[10];
  const float* up_w1    = (const float*)d_in[11];
  const float* up_b1    = (const float*)d_in[12];
  const float* up_w2    = (const float*)d_in[13];
  const float* up_b2    = (const float*)d_in[14];
  const float* cls_w1   = (const float*)d_in[15];
  const float* cls_b1   = (const float*)d_in[16];
  const float* cls_w2   = (const float*)d_in[17];
  const float* cls_b2   = (const float*)d_in[18];
  float* out = (float*)d_out;

  const int Nn = 50000, H = 256, E = 80000, T = 3, L = 2, Cc = 16;

  float* ws   = (float*)d_ws;
  float* hA   = ws;                          // [N,H]
  float* hB   = hA + (size_t)Nn * H;         // [N,H]
  float* agg  = hB + (size_t)Nn * H;         // [N,H]
  float* mbuf = agg + (size_t)Nn * H;        // [E,H], reused as tmp [N,H]
  float* tmp  = mbuf;

  // bf16 transposed-weight region after the fp32 buffers (16B aligned).
  unsigned short* wt = (unsigned short*)(mbuf + (size_t)E * H);
  unsigned short* wt_in = wt;                 wt += (size_t)H * H;
  unsigned short* wt_m1[3];
  unsigned short* wt_m2[3];
  for (int t = 0; t < 3; ++t) { wt_m1[t] = wt; wt += (size_t)H * H; }
  for (int t = 0; t < 3; ++t) { wt_m2[t] = wt; wt += (size_t)H * H; }
  unsigned short* wt_u1[2];
  unsigned short* wt_u2[2];
  for (int l = 0; l < 2; ++l) { wt_u1[l] = wt; wt += (size_t)2 * H * H; }
  for (int l = 0; l < 2; ++l) { wt_u2[l] = wt; wt += (size_t)H * H; }
  unsigned short* wt_c1 = wt;                 wt += (size_t)H * H;
  unsigned short* wt_c2 = wt;

  dim3 blk(256, 1, 1);
  auto mkgrid = [](int M, int Np) {
    return dim3((unsigned)((Np + BN - 1) / BN), (unsigned)((M + BM - 1) / BM), 1);
  };
  auto convT = [&](const float* src, unsigned short* dst, int K, int N) {
    int total = K * N;
    conv_transpose_bf16<<<dim3((unsigned)((total + 255) / 256)), blk, 0, stream>>>(
        src, dst, K, N);
  };

  // ---- pre-pass: weights -> transposed bf16 ----
  convT(in_w, wt_in, H, H);
  for (int t = 0; t < T; ++t) {
    convT(msg_w1 + (size_t)t * H * H, wt_m1[t], H, H);
    convT(msg_w2 + (size_t)t * H * H, wt_m2[t], H, H);
  }
  for (int l = 0; l < L; ++l) {
    convT(up_w1 + (size_t)l * 2 * H * H, wt_u1[l], 2 * H, H);
    convT(up_w2 + (size_t)l * H * H,     wt_u2[l], H, H);
  }
  convT(cls_w1, wt_c1, H, H);
  convT(cls_w2, wt_c2, H, Cc);

  // h = relu((x * z_feat) @ in_proj_w + b)
  gnn_wmma_gemm<<<mkgrid(Nn, H), blk, 0, stream>>>(
      x, H, nullptr, 0, H, nullptr, feat_la, wt_in, in_b, nullptr,
      hA, H, nullptr, Nn, H, H, 1);

  float* hcur = hA;
  float* hnext = hB;
  for (int l = 0; l < L; ++l) {
    zero_f32<<<dim3(512), blk, 0, stream>>>(agg, (long)Nn * H);
    for (int t = 0; t < T; ++t) {
      // m1 = relu(h[edge_src[t]] @ msg_w1[t] + b1[t])
      gnn_wmma_gemm<<<mkgrid(E, H), blk, 0, stream>>>(
          hcur, H, nullptr, 0, H, edge_src + (size_t)t * E, nullptr,
          wt_m1[t], msg_b1 + (size_t)t * H, nullptr,
          mbuf, H, nullptr, E, H, H, 1);
      // agg[dst] += z_edge[t] * (m1 @ msg_w2[t] + b2[t])  (fused scatter-add)
      gnn_wmma_gemm<<<mkgrid(E, H), blk, 0, stream>>>(
          mbuf, H, nullptr, 0, H, nullptr, nullptr,
          wt_m2[t], msg_b2 + (size_t)t * H, edge_la + t,
          agg, H, edge_dst + (size_t)t * E, E, H, H, 0);
    }
    // tmp = relu(concat([h, agg]) @ up_w1[l] + b1[l])   (K-split, no concat buf)
    gnn_wmma_gemm<<<mkgrid(Nn, H), blk, 0, stream>>>(
        hcur, H, agg, H, H, nullptr, nullptr,
        wt_u1[l], up_b1 + (size_t)l * H, nullptr,
        tmp, H, nullptr, Nn, H, 2 * H, 1);
    // h = tmp @ up_w2[l] + b2[l]
    gnn_wmma_gemm<<<mkgrid(Nn, H), blk, 0, stream>>>(
        tmp, H, nullptr, 0, H, nullptr, nullptr,
        wt_u2[l], up_b2 + (size_t)l * H, nullptr,
        hnext, H, nullptr, Nn, H, H, 0);
    float* sw = hcur; hcur = hnext; hnext = sw;
  }

  // logits = relu(h @ cls_w1 + b1) @ cls_w2 + b2
  gnn_wmma_gemm<<<mkgrid(Nn, H), blk, 0, stream>>>(
      hcur, H, nullptr, 0, H, nullptr, nullptr,
      wt_c1, cls_b1, nullptr, tmp, H, nullptr, Nn, H, H, 1);
  gnn_wmma_gemm<<<mkgrid(Nn, Cc), blk, 0, stream>>>(
      tmp, H, nullptr, 0, H, nullptr, nullptr,
      wt_c2, cls_b2, nullptr, out, Cc, nullptr, Nn, Cc, H, 0);
}
